// GroupColumnParallelLinearDeepSeekV3_22153441312855
// MI455X (gfx1250) — 8-bit matrix operand order repaired and checked by reading, compiled, not run on hardware
//
#include <hip/hip_runtime.h>

#define G_CT   8
#define T_CT   1024
#define K_CT   7168
#define N_CT   2048
#define KB_CT  (K_CT / 128)
#define NB_CT  (N_CT / 128)

typedef __attribute__((ext_vector_type(16))) int   v16i;
typedef __attribute__((ext_vector_type(8)))  float v8f;
typedef __attribute__((ext_vector_type(4)))  float v4f;
typedef __attribute__((ext_vector_type(4)))  int   v4i;
typedef __attribute__((ext_vector_type(2)))  int   v2i;

typedef __attribute__((address_space(1))) v4i GV4i;
typedef __attribute__((address_space(3))) v4i LV4i;

#if __has_builtin(__builtin_amdgcn_global_load_async_to_lds_b128)
#define USE_ASYNC_LDS 1
#else
#define USE_ASYNC_LDS 0
#endif

__device__ __forceinline__ void wait_asynccnt0() {
#if __has_builtin(__builtin_amdgcn_s_wait_asynccnt)
  __builtin_amdgcn_s_wait_asynccnt(0);
#else
  asm volatile("s_wait_asynccnt 0x0" ::: "memory");
#endif
}

#define XQ_BYTES ((size_t)G_CT * T_CT * K_CT)
#define AS_BYTES ((size_t)G_CT * T_CT * KB_CT * 4)
#define WQ_OFF   (XQ_BYTES + AS_BYTES)

__device__ __forceinline__ unsigned int f32_to_e4m3(float x) {
  unsigned int u    = __float_as_uint(x);
  unsigned int sign = (u >> 24) & 0x80u;
  unsigned int a    = u & 0x7FFFFFFFu;
  if (a >= 0x7F800000u) return sign | 0x7Fu;
  if (a < 0x3C800000u) {
    int m = __float2int_rn(__uint_as_float(a) * 512.0f);
    return sign | (unsigned int)m;
  }
  unsigned int r = a + 0x0007FFFFu + ((a >> 20) & 1u);
  unsigned int e = (r >> 23);
  unsigned int code = ((e - 120u) << 3) | ((r >> 20) & 7u);
  if (code > 0x7Eu) code = 0x7Eu;
  return sign | code;
}

__global__ __launch_bounds__(256) void act_quant_kernel(
    const float* __restrict__ xs,
    unsigned char* __restrict__ xq,
    float* __restrict__ ascale) {
  int w    = blockIdx.x * 8 + (threadIdx.x >> 5);
  int lane = threadIdx.x & 31;
  int kb = w % KB_CT;
  int gt = w / KB_CT;

  const v4f* src = (const v4f*)(xs + (size_t)gt * K_CT + (size_t)kb * 128);
  v4f v = __builtin_nontemporal_load(&src[lane]);

  float m = fmaxf(fmaxf(fabsf(v[0]), fabsf(v[1])), fmaxf(fabsf(v[2]), fabsf(v[3])));
  #pragma unroll
  for (int off = 16; off; off >>= 1) m = fmaxf(m, __shfl_xor(m, off, 32));

  float inv = (m > 0.0f) ? (448.0f / m) : 0.0f;
  unsigned int b =  f32_to_e4m3(v[0] * inv)
                 | (f32_to_e4m3(v[1] * inv) << 8)
                 | (f32_to_e4m3(v[2] * inv) << 16)
                 | (f32_to_e4m3(v[3] * inv) << 24);
  ((unsigned int*)xq)[(size_t)gt * (K_CT / 4) + kb * 32 + lane] = b;
  if (lane == 0) ascale[(size_t)gt * KB_CT + kb] = m * (1.0f / 448.0f);
}

__global__ __launch_bounds__(256) void w_quant_kernel(
    const float* __restrict__ w,
    unsigned char* __restrict__ wq) {
  size_t i = (size_t)blockIdx.x * 256 + threadIdx.x;
  const v4f* src = (const v4f*)w;
  v4f v = __builtin_nontemporal_load(&src[i]);
  unsigned int b =  f32_to_e4m3(v[0])
                 | (f32_to_e4m3(v[1]) << 8)
                 | (f32_to_e4m3(v[2]) << 16)
                 | (f32_to_e4m3(v[3]) << 24);
  ((unsigned int*)wq)[i] = b;
}

__global__ __launch_bounds__(256) void gemm_fp8_kernel(
    const unsigned char* __restrict__ xq,
    const float* __restrict__ ascale,
    const unsigned char* __restrict__ wq,
    const float* __restrict__ wscale,
    const float* __restrict__ bias,
    float* __restrict__ out) {
  __shared__ __align__(16) unsigned char Bsh[2][128 * 128];

  const int g    = blockIdx.z;
  const int nb   = blockIdx.x;
  const int n0   = nb * 128;
  const int t0   = blockIdx.y * 128;
  const int wid  = threadIdx.x >> 5;
  const int lane = threadIdx.x & 31;
  const int row  = lane & 15;
  const int half = lane >> 4;
  const int tw   = t0 + wid * 16;

  const unsigned char* Abase = xq + ((size_t)(g * T_CT + tw + row)) * K_CT;
  const unsigned char* Wbase = wq + ((size_t)(g * N_CT + n0)) * K_CT;

  v8f acc[8];
  #pragma unroll
  for (int nt = 0; nt < 8; ++nt) acc[nt] = {};

  auto stage = [&](int buf, int kb) {
    #pragma unroll
    for (int i = 0; i < 4; ++i) {
      int c  = (int)threadIdx.x * 4 + i;
      int r  = c >> 3;
      int cc = (c & 7) * 16;
      const unsigned char* gp = Wbase + (size_t)r * K_CT + (size_t)kb * 128 + cc;
      unsigned char*       lp = &Bsh[buf][r * 128 + cc];
#if USE_ASYNC_LDS
      __builtin_amdgcn_global_load_async_to_lds_b128(
          (GV4i*)(uintptr_t)gp, (LV4i*)(unsigned int)(uintptr_t)lp, 0, 0);
#else
      *(v4i*)lp = *(const v4i*)gp;
#endif
    }
  };

  stage(0, 0);
#if USE_ASYNC_LDS
  wait_asynccnt0();
#endif
  __syncthreads();

  for (int kb = 0; kb < KB_CT; ++kb) {
    const int buf = kb & 1;
    if (kb + 1 < KB_CT) stage(buf ^ 1, kb + 1);

    const float ws = wscale[((size_t)g * NB_CT + nb) * KB_CT + kb];
    float as[8];
    const float* ap = ascale + ((size_t)(g * T_CT + tw + half * 8)) * KB_CT + kb;
    #pragma unroll
    for (int j = 0; j < 8; ++j) as[j] = ap[(size_t)j * KB_CT] * ws;

    v16i A;
    const unsigned char* arow = Abase + (size_t)kb * 128 + half * 8;
    #pragma unroll
    for (int c = 0; c < 8; ++c) {
      v2i t2 = *(const v2i*)(arow + c * 16);
      A[2 * c]     = t2[0];
      A[2 * c + 1] = t2[1];
    }

    #pragma unroll
    for (int nt = 0; nt < 8; ++nt) {
      v16i B;
      const unsigned char* brow = &Bsh[buf][(nt * 16 + row) * 128 + half * 8];
      #pragma unroll
      for (int c = 0; c < 8; ++c) {
        v2i t2 = *(const v2i*)(brow + c * 16);
        B[2 * c]     = t2[0];
        B[2 * c + 1] = t2[1];
      }
      v8f z = {};
      v8f d = __builtin_amdgcn_wmma_f32_16x16x128_fp8_fp8(A, B, (short)0, z,
                                                          false, false);
      #pragma unroll
      for (int j = 0; j < 8; ++j) acc[nt][j] = fmaf(as[j], d[j], acc[nt][j]);
    }

#if USE_ASYNC_LDS
    wait_asynccnt0();
#endif
    __syncthreads();
  }

  #pragma unroll
  for (int nt = 0; nt < 8; ++nt) {
    int n = n0 + nt * 16 + row;
    float bz = bias[(size_t)g * N_CT + n];
    #pragma unroll
    for (int j = 0; j < 8; ++j) {
      int t = tw + half * 8 + j;
      out[((size_t)(g * T_CT + t)) * N_CT + n] = acc[nt][j] + bz;
    }
  }
}

extern "C" void kernel_launch(void* const* d_in, const int* in_sizes, int n_in,
                              void* d_out, int out_size, void* d_ws, size_t ws_size,
                              hipStream_t stream) {
  const float* xs     = (const float*)d_in[0];
  const float* weight = (const float*)d_in[1];
  const float* wscale = (const float*)d_in[2];
  const float* bias   = (const float*)d_in[3];
  float* out          = (float*)d_out;

  unsigned char* ws = (unsigned char*)d_ws;
  unsigned char* xq = ws;
  float*         as = (float*)(ws + XQ_BYTES);
  unsigned char* wq = ws + WQ_OFF;

  act_quant_kernel<<<(G_CT * T_CT * KB_CT) / 8, 256, 0, stream>>>(xs, xq, as);

  w_quant_kernel<<<(int)(((size_t)G_CT * N_CT * K_CT / 4) / 256), 256, 0, stream>>>(weight, wq);

  dim3 grid(N_CT / 128, T_CT / 128, G_CT);
  gemm_fp8_kernel<<<grid, 256, 0, stream>>>(xq, as, wq, wscale, bias, out);
}
